// EnhancedAttention_37211596653065
// MI455X (gfx1250) — matrix operand order repaired and checked by reading, not compiled, not run on hardware
//
#include <hip/hip_runtime.h>

// ---------------------------------------------------------------------------
// EnhancedAttention for MI455X (gfx1250): bf16 WMMA pipeline.
//   B=8, S=1024, D_MODEL=1024, H=16, DK=64, SCALE=16.0
// Output tuple: (output [8,1024,1024], attn [8,16,1024,1024]) concat in d_out.
// ---------------------------------------------------------------------------

typedef __attribute__((ext_vector_type(16))) __bf16        v16bf;
typedef __attribute__((ext_vector_type(8)))  float         v8f;
typedef __attribute__((ext_vector_type(4)))  unsigned int  u32x4;
typedef __attribute__((ext_vector_type(2)))  unsigned int  u32x2;
typedef __attribute__((ext_vector_type(4)))  float         f32x4;

#define NH   16
#define SEQ  1024
#define DK   64
#define DM   1024
#define NB   8

// float -> bf16 with round-to-nearest-even (bit twiddle; inputs are finite)
__device__ __forceinline__ unsigned short f2bf(float f) {
    unsigned int x = __float_as_uint(f);
    x += 0x7FFFu + ((x >> 16) & 1u);
    return (unsigned short)(x >> 16);
}

// Two 16-byte loads -> one 16xbf16 WMMA fragment (per-lane); works for
// global (global_load_b128) and LDS (ds_load_b128) source pointers.
__device__ __forceinline__ v16bf load_frag2(const unsigned short* p0,
                                            const unsigned short* p1) {
    union { u32x4 u[2]; v16bf v; } f;
    f.u[0] = *(const u32x4*)p0;
    f.u[1] = *(const u32x4*)p1;
    return f.v;
}

// ---------------------------------------------------------------------------
// Kernel 1: elementwise f32 -> bf16 (RNE)
// ---------------------------------------------------------------------------
__global__ void cvt_f32_bf16(const float* __restrict__ in,
                             unsigned short* __restrict__ out, int n) {
    int i = blockIdx.x * blockDim.x + threadIdx.x;
    int stride = gridDim.x * blockDim.x;
    for (; i < n; i += stride) out[i] = f2bf(in[i]);
}

// ---------------------------------------------------------------------------
// Kernel 2: fused Q/K/V projection.  out = x @ W^T  (A·B^T, both K-contiguous)
// One 16x16 output tile per wave; K=1024 in 32 steps of wmma_f32_16x16x32_bf16.
// Q,K stored [bh][s][dk]; V stored transposed [bh][dk][s].
// ---------------------------------------------------------------------------
__global__ void proj_qkv(const unsigned short* __restrict__ xq,
                         const unsigned short* __restrict__ xk,
                         const unsigned short* __restrict__ xv,
                         const unsigned short* __restrict__ wq,
                         const unsigned short* __restrict__ wk,
                         const unsigned short* __restrict__ wv,
                         unsigned short* __restrict__ Qh,
                         unsigned short* __restrict__ Kh,
                         unsigned short* __restrict__ Vt) {
    const int lane = threadIdx.x & 31;
    const int wid  = threadIdx.x >> 5;
    const int flat = blockIdx.x * (blockDim.x >> 5) + wid;
    const int mat  = flat >> 15;        // 0=Q 1=K 2=V (512*64 = 32768 tiles each)
    const int rem  = flat & 32767;
    const int mt   = rem >> 6;          // token tile [0,512)
    const int nt   = rem & 63;          // d_model tile [0,64)

    const unsigned short* A = (mat == 0) ? xq : (mat == 1) ? xk : xv;
    const unsigned short* W = (mat == 0) ? wq : (mat == 1) ? wk : wv;

    const int am   = mt * 16 + (lane & 15);     // A row (token)
    const int wn   = nt * 16 + (lane & 15);     // B row (= output column, W row)
    const int aoff = (lane >> 4) * 8;           // A-frag K sub-chunk select
    const int boff = (lane >> 4) * 8;           // B-frag K sub-chunk select

    v8f acc = {};
    for (int k0 = 0; k0 < DM; k0 += 32) {
        const unsigned short* ap = A + (size_t)am * DM + k0 + aoff;
        const unsigned short* wp = W + (size_t)wn * DM + k0 + boff;
        __builtin_prefetch(ap + 64, 0, 0);      // global_prefetch_b8 next k-step
        v16bf a = load_frag2(ap, ap + 16);      // lane<16: K{0..7,16..23}, lane>=16: K{8..15,24..31}
        v16bf b = load_frag2(wp, wp + 16);      // lane<16: K{0..7,16..23}, lane>=16: K{8..15,24..31}
        acc = __builtin_amdgcn_wmma_f32_16x16x32_bf16(
                  false, a, false, b, (short)0, acc, false, false);
    }

    // C layout: VGPR r, lanes 0-15 -> M=r, lanes 16-31 -> M=r+8; N = lane&15
    const int n  = nt * 16 + (lane & 15);
    const int h  = n >> 6, d = n & 63;
    const int mh = (lane >> 4) << 3;
#pragma unroll
    for (int r = 0; r < 8; r++) {
        int tok = mt * 16 + r + mh;
        int b_  = tok >> 10, s_ = tok & 1023;
        unsigned short bv = f2bf(acc[r]);
        if (mat == 2) {
            Vt[((size_t)(b_ * NH + h) * DK + d) * SEQ + s_] = bv;       // transposed
        } else {
            unsigned short* D = (mat == 0) ? Qh : Kh;
            D[((size_t)(b_ * NH + h) * SEQ + s_) * DK + d] = bv;
        }
    }
}

// ---------------------------------------------------------------------------
// Kernel 3: attention. One block (4 waves, 128 thr) per (bh, 16-row q tile).
// Dynamic LDS layout (bytes):
//   [0,      65536)  sc   : f32 exp'd scores [16][1024]
//   [65536,  66048)  red  : f32 [128] partial row sums
//   [66048,  66112)  invs : f32 [16] 1/rowsum
//   [66112,  98880)  scb  : bf16 normalized attn [16][1024]  (A frags, phase 4)
//   [98880, 100928)  qtile: bf16 Q tile [16][64]  (async-staged from global)
// ---------------------------------------------------------------------------
__global__ void attn_kernel(const unsigned short* __restrict__ Qh,
                            const unsigned short* __restrict__ Kh,
                            const unsigned short* __restrict__ Vt,
                            float* __restrict__ attn_out,
                            unsigned short* __restrict__ ctx) {
    extern __shared__ float lds[];
    float*          sc    = lds;                              // [16][1024]
    float*          red   = lds + 16 * SEQ;                   // [128]
    float*          invs  = red + 128;                        // [16]
    unsigned short* scb   = (unsigned short*)(invs + 16);     // [16][1024] bf16
    unsigned short* qtile = scb + 16 * SEQ;                   // [16][64]   bf16

    const int lane = threadIdx.x & 31;
    const int wid  = threadIdx.x >> 5;
    const int tid  = threadIdx.x;
    const int bh   = blockIdx.x >> 6;   // [0,128)
    const int qt   = blockIdx.x & 63;   // q tile
    const int q0   = qt * 16;

    const unsigned short* Qb = Qh + (size_t)bh * SEQ * DK;
    const unsigned short* Kb = Kh + (size_t)bh * SEQ * DK;
    const unsigned short* Vb = Vt + (size_t)bh * DK * SEQ;

    // Stage the 2KB contiguous Q tile (16 rows x 64 bf16) into LDS with the
    // CDNA5 async global->LDS mover (tracked by ASYNCcnt). Wave 0 issues
    // 4 x (32 lanes x b128) = 2KB.
    if (wid == 0) {
        const unsigned qbase = (unsigned)(size_t)qtile;       // LDS byte offset
#pragma unroll
        for (int i = 0; i < 4; i++) {
            const int chunk = i * 32 + lane;                  // 16B chunks
            const unsigned short* gp = Qb + (size_t)q0 * DK + chunk * 8;
            const unsigned       lo  = qbase + chunk * 16;
            asm volatile("global_load_async_to_lds_b128 %0, %1, off"
                         :: "v"(lo), "v"(gp) : "memory");
        }
    }

    // phase 0: zero f32 score buffer (fully-masked key tiles stay 0)
    {
        f32x4 z = {};
        f32x4* p = (f32x4*)sc;
        for (int i = tid; i < 16 * SEQ / 4; i += blockDim.x) p[i] = z;
    }
    if (wid == 0) {
        asm volatile("s_wait_asynccnt 0x0" ::: "memory");     // Q tile landed
    }
    __syncthreads();

    // Per-lane fragment geometry
    const int aoff = (lane >> 4) * 8;
    const int boff = (lane >> 4) * 8;
    const int cm   = (lane >> 4) << 3;
    const int cn   = lane & 15;

    // Q A-fragments for d=0..31 and d=32..63, from LDS (ds_load_b128)
    const unsigned short* qp = qtile + (lane & 15) * DK + aoff;
    const v16bf aQ0 = load_frag2(qp,      qp + 16);
    const v16bf aQ1 = load_frag2(qp + 32, qp + 48);

    // phase 1: scores = exp(clip(QK^T/16)), causal; waves split key tiles
    for (int kt = wid; kt <= qt; kt += 4) {
        const int k0 = kt * 16;
        const unsigned short* kp = Kb + (size_t)(k0 + (lane & 15)) * DK + boff;
        v16bf bK0 = load_frag2(kp,      kp + 16);
        v16bf bK1 = load_frag2(kp + 32, kp + 48);
        v8f c = {};
        c = __builtin_amdgcn_wmma_f32_16x16x32_bf16(false, aQ0, false, bK0, (short)0, c, false, false);
        c = __builtin_amdgcn_wmma_f32_16x16x32_bf16(false, aQ1, false, bK1, (short)0, c, false, false);
        const bool diag = (kt == qt);
#pragma unroll
        for (int r = 0; r < 8; r++) {
            int m = r + cm;
            float v = c[r] * (1.0f / 16.0f);               // /SCALE
            v = fminf(30.0f, fmaxf(-30.0f, v));            // clip
            float e = __expf(v);                           // no max-sub needed (clipped)
            if (diag && cn > m) e = 0.0f;                  // causal mask -> exp(-1e9)=0
            sc[m * SEQ + k0 + cn] = e;
        }
    }
    __syncthreads();

    // phase 2: row sums -> 1/sum (diagonal always unmasked => sum > 0)
    {
        int row = tid >> 3, seg = (tid & 7) * 128;
        float s = 0.f;
        const float* p = sc + row * SEQ + seg;
        for (int j = 0; j < 128; j++) s += p[j];
        red[tid] = s;
    }
    __syncthreads();
    if (tid < 16) {
        float s = 0.f;
        for (int j = 0; j < 8; j++) s += red[tid * 8 + j];
        invs[tid] = 1.0f / s;
    }
    __syncthreads();

    // phase 3: normalize; stream attn (f32) to global AND keep a packed bf16
    // copy in LDS for the phase-4 A fragments.
    {
        int row = tid >> 3, seg = (tid & 7) * 128;
        float iv = invs[row];
        const float* p = sc + row * SEQ + seg;
        float* g = attn_out + ((size_t)bh * SEQ + (q0 + row)) * SEQ + seg;
        unsigned* pb = (unsigned*)(scb + row * SEQ + seg);
        for (int j = 0; j < 128; j += 4) {
            f32x4 v = *(const f32x4*)(p + j);
            v *= iv;
            *(f32x4*)(g + j) = v;
            u32x2 packed;
            packed.x = (unsigned)f2bf(v.x) | ((unsigned)f2bf(v.y) << 16);
            packed.y = (unsigned)f2bf(v.z) | ((unsigned)f2bf(v.w) << 16);
            *(u32x2*)(pb + (j >> 1)) = packed;
        }
    }
    __syncthreads();

    // phase 4: context = attn @ V; wave wid owns d-columns [wid*16, wid*16+16)
    {
        const int dcol = wid * 16 + (lane & 15);
        const int kend = ((q0 + 16) + 31) & ~31;           // causal bound, rounded to 32
        v8f acc = {};
        for (int ks = 0; ks < kend; ks += 32) {
            const unsigned short* ap = scb + (lane & 15) * SEQ + ks + aoff;
            v16bf a = load_frag2(ap, ap + 16);             // ds_load_b128 x2
            const unsigned short* vp = Vb + (size_t)dcol * SEQ + ks + boff;
            v16bf b = load_frag2(vp, vp + 16);             // V^T rows are contiguous
            acc = __builtin_amdgcn_wmma_f32_16x16x32_bf16(
                      false, a, false, b, (short)0, acc, false, false);
        }
        const int b_ = bh >> 4, h_ = bh & 15;
#pragma unroll
        for (int r = 0; r < 8; r++) {
            int m = r + cm;
            ctx[(size_t)(b_ * SEQ + q0 + m) * DM + h_ * DK + wid * 16 + cn] = f2bf(acc[r]);
        }
    }
}

// ---------------------------------------------------------------------------
// Kernel 4: output projection  out = ctx @ Wo^T   (f32 result to d_out)
// ---------------------------------------------------------------------------
__global__ void out_proj(const unsigned short* __restrict__ ctxb,
                         const unsigned short* __restrict__ wo,
                         float* __restrict__ out) {
    const int lane = threadIdx.x & 31;
    const int wid  = threadIdx.x >> 5;
    const int flat = blockIdx.x * (blockDim.x >> 5) + wid;
    const int mt   = flat >> 6;
    const int nt   = flat & 63;
    const int am   = mt * 16 + (lane & 15);
    const int wn   = nt * 16 + (lane & 15);
    const int aoff = (lane >> 4) * 8;
    const int boff = (lane >> 4) * 8;

    v8f acc = {};
    for (int k0 = 0; k0 < DM; k0 += 32) {
        const unsigned short* ap = ctxb + (size_t)am * DM + k0 + aoff;
        const unsigned short* wp = wo   + (size_t)wn * DM + k0 + boff;
        __builtin_prefetch(wp + 64, 0, 0);
        v16bf a = load_frag2(ap, ap + 16);
        v16bf b = load_frag2(wp, wp + 16);
        acc = __builtin_amdgcn_wmma_f32_16x16x32_bf16(
                  false, a, false, b, (short)0, acc, false, false);
    }
    const int n  = nt * 16 + (lane & 15);
    const int mh = (lane >> 4) << 3;
#pragma unroll
    for (int r = 0; r < 8; r++) {
        out[(size_t)(mt * 16 + r + mh) * DM + n] = acc[r];
    }
}

// ---------------------------------------------------------------------------
extern "C" void kernel_launch(void* const* d_in, const int* in_sizes, int n_in,
                              void* d_out, int out_size, void* d_ws, size_t ws_size,
                              hipStream_t stream) {
    const float* q  = (const float*)d_in[0];
    const float* k  = (const float*)d_in[1];
    const float* v  = (const float*)d_in[2];
    // d_in[3] = mask (tril, int32) — causal structure applied analytically
    const float* Wq = (const float*)d_in[4];
    const float* Wk = (const float*)d_in[5];
    const float* Wv = (const float*)d_in[6];
    const float* Wo = (const float*)d_in[7];

    const size_t MEG = 1024 * 1024;
    unsigned short* u   = (unsigned short*)d_ws;   // offsets in bf16 elements
    unsigned short* xq  = u;                       //  8M
    unsigned short* xk  = u + 8  * MEG;            //  8M
    unsigned short* xv  = u + 16 * MEG;            //  8M
    unsigned short* wqb = u + 24 * MEG;            //  1M
    unsigned short* wkb = u + 25 * MEG;
    unsigned short* wvb = u + 26 * MEG;
    unsigned short* wob = u + 27 * MEG;
    unsigned short* Qh  = u + 28 * MEG;            //  8M  [bh][s][dk]
    unsigned short* Kh  = u + 36 * MEG;            //  8M  [bh][s][dk]
    unsigned short* Vt  = u + 44 * MEG;            //  8M  [bh][dk][s]
    unsigned short* ctx = u + 52 * MEG;            //  8M  [b,s][d_model]
    // total 60M bf16 = 120 MB of workspace

    float* out_proj_dst = (float*)d_out;                   // [8,1024,1024]
    float* attn_dst     = (float*)d_out + 8 * MEG;         // [8,16,1024,1024]

    // 1) convert inputs + weights to bf16
    const int nx = NB * SEQ * DM;       // 8388608
    const int nw = DM * DM;             // 1048576
    cvt_f32_bf16<<<dim3(2048), dim3(256), 0, stream>>>(q,  xq,  nx);
    cvt_f32_bf16<<<dim3(2048), dim3(256), 0, stream>>>(k,  xk,  nx);
    cvt_f32_bf16<<<dim3(2048), dim3(256), 0, stream>>>(v,  xv,  nx);
    cvt_f32_bf16<<<dim3(1024), dim3(256), 0, stream>>>(Wq, wqb, nw);
    cvt_f32_bf16<<<dim3(1024), dim3(256), 0, stream>>>(Wk, wkb, nw);
    cvt_f32_bf16<<<dim3(1024), dim3(256), 0, stream>>>(Wv, wvb, nw);
    cvt_f32_bf16<<<dim3(1024), dim3(256), 0, stream>>>(Wo, wob, nw);

    // 2) Q/K/V projections: 3 * (512 m-tiles * 64 n-tiles) waves / 4 per block
    proj_qkv<<<dim3(24576), dim3(128), 0, stream>>>(xq, xk, xv, wqb, wkb, wvb,
                                                    Qh, Kh, Vt);

    // 3) attention: 128 (b,h) * 64 q-tiles blocks; ~100928 B dynamic LDS
    const size_t lds_bytes = (16 * SEQ + 128 + 16) * sizeof(float)   // sc/red/invs
                           + 16 * SEQ * sizeof(unsigned short)       // scb
                           + 16 * DK * sizeof(unsigned short);       // qtile
    attn_kernel<<<dim3(NB * NH * 64), dim3(128), lds_bytes, stream>>>(
        Qh, Kh, Vt, attn_dst, ctx);

    // 4) output projection: 512*64 waves / 4 per block
    out_proj<<<dim3(8192), dim3(128), 0, stream>>>(ctx, wob, out_proj_dst);
}
